// CameraContrast_32083405701138
// MI455X (gfx1250) — matrix operand order repaired and checked by reading, not compiled, not run on hardware
//
#include <hip/hip_runtime.h>
#include <math.h>

// ---------------------------------------------------------------------------
// CameraContrast loss for MI455X (gfx1250, wave32, WMMA).
//   sims = (features/||features||) @ proxy^T / TEMP        [B,P] via bf16 WMMA
//   per-row: top-50 hard negatives + logsumexp(pos ∪ hardnegs) - mean(pos)
//
// v3b: double-buffered 128x128 WMMA GEMM; LDS staging uses CDNA5
// GLOBAL_LOAD_ASYNC_TO_LDS_B128 (ASYNCcnt) -- builtin takes int4-vector
// pointers in AS1 (global) / AS3 (LDS). Fallback paths preserved.
// ---------------------------------------------------------------------------

typedef __attribute__((ext_vector_type(16))) __bf16          v16bf;
typedef __attribute__((ext_vector_type(8)))  float           v8f;
typedef __attribute__((ext_vector_type(8)))  unsigned short  ushort8;
typedef __attribute__((ext_vector_type(16))) unsigned short  ushort16;

#define TILE_K 32
#define LDS_STRIDE 40   // 40 halves = 80B rows: 16B aligned, conflict-free strides
#define NUM_HARDS 50
#define NEG_INF (-3.0e38f)

#if __has_builtin(__builtin_amdgcn_global_load_async_to_lds_b128) && \
    __has_builtin(__builtin_amdgcn_s_wait_asynccnt)
#define HAVE_ASYNC_LDS 1
#else
#define HAVE_ASYNC_LDS 0
#endif

// The async-to-LDS builtins want pointers to a 16B generic int vector,
// in address space 1 (global, printed as __device__) and 3 (LDS).
typedef int int4g __attribute__((vector_size(16)));
typedef __attribute__((address_space(1))) int4g as1_int4;
typedef __attribute__((address_space(3))) int4g as3_int4;

__device__ __forceinline__ as1_int4* to_global(const void* p) {
    return (as1_int4*)(unsigned long long)p;
}
__device__ __forceinline__ as3_int4* to_lds(const void* p) {
    return (as3_int4*)(unsigned)(unsigned long long)p;
}

__device__ __forceinline__ unsigned short f32_to_bf16(float f) {
    // round-to-nearest-even
    unsigned int u = __float_as_uint(f);
    unsigned int r = (u + 0x7fffu + ((u >> 16) & 1u)) >> 16;
    return (unsigned short)r;
}

__device__ __forceinline__ v16bf cat_bf16(ushort8 lo, ushort8 hi) {
    const ushort16 cat = __builtin_shufflevector(lo, hi,
        0,1,2,3,4,5,6,7,8,9,10,11,12,13,14,15);
    return __builtin_bit_cast(v16bf, cat);
}

// ---------------------------------------------------------------------------
// Kernel 1: per-row inverse L2 norm of features
// ---------------------------------------------------------------------------
__global__ __launch_bounds__(256) void invnorm_kernel(
    const float* __restrict__ feat, float* __restrict__ invn, int D)
{
    __shared__ float s[256];
    const int b = blockIdx.x, tid = threadIdx.x;
    const float* r = feat + (size_t)b * D;
    float a = 0.f;
    for (int j = tid; j < D; j += 256) { float v = r[j]; a += v * v; }
    s[tid] = a; __syncthreads();
    for (int st = 128; st > 0; st >>= 1) {
        if (tid < st) s[tid] += s[tid + st];
        __syncthreads();
    }
    if (tid == 0) invn[b] = 1.0f / sqrtf(s[0]);
}

// ---------------------------------------------------------------------------
// Kernel 1b/1c: one-time f32 -> bf16 conversion passes
// ---------------------------------------------------------------------------
__global__ __launch_bounds__(256) void convA_kernel(
    const float* __restrict__ feat, const float* __restrict__ invn,
    unsigned short* __restrict__ fA, int D)
{
    const int b = blockIdx.x, tid = threadIdx.x;
    const float s = invn[b];
    const float* src = feat + (size_t)b * D;
    unsigned short* dst = fA + (size_t)b * D;
    for (int j = tid * 4; j < D; j += 256 * 4) {
        const float4 v = *(const float4*)&src[j];
        ushort4 o;
        o.x = f32_to_bf16(v.x * s); o.y = f32_to_bf16(v.y * s);
        o.z = f32_to_bf16(v.z * s); o.w = f32_to_bf16(v.w * s);
        *(ushort4*)&dst[j] = o;
    }
}

__global__ __launch_bounds__(256) void convB_kernel(
    const float* __restrict__ proxy, unsigned short* __restrict__ pB, long n4)
{
    long i = (long)blockIdx.x * blockDim.x + threadIdx.x;
    const long stride = (long)gridDim.x * blockDim.x;
    for (; i < n4; i += stride) {
        const float4 v = ((const float4*)proxy)[i];
        ushort4 o;
        o.x = f32_to_bf16(v.x); o.y = f32_to_bf16(v.y);
        o.z = f32_to_bf16(v.z); o.w = f32_to_bf16(v.w);
        ((ushort4*)pB)[i] = o;
    }
}

// ---------------------------------------------------------------------------
// Kernel 2 (main path): sims = fA @ pB^T * (1/TEMP), bf16 operands in memory.
// Block = 256 threads (8 waves). Block tile 128x128, wave tile 32x64
// (2x4 WMMA accumulators). Double-buffered LDS staging via async-to-LDS.
// ---------------------------------------------------------------------------
__device__ __forceinline__ void stage_tiles(
    const unsigned short* __restrict__ fA, const unsigned short* __restrict__ pB,
    unsigned short* __restrict__ lA, unsigned short* __restrict__ lB,
    int m0, int n0, int k0, int D, int tid)
{
#if HAVE_ASYNC_LDS
    // memory -> LDS directly, 16B per lane per op, tracked on ASYNCcnt
    #pragma unroll
    for (int c = tid; c < 512; c += 256) {
        const int row = c >> 2;
        const int kc  = (c & 3) << 3;  // halves
        __builtin_amdgcn_global_load_async_to_lds_b128(
            to_global(&fA[(size_t)(m0 + row) * D + k0 + kc]),
            to_lds(&lA[row * LDS_STRIDE + kc]), 0, 0);
    }
    #pragma unroll
    for (int c = tid; c < 512; c += 256) {
        const int row = c >> 2;
        const int kc  = (c & 3) << 3;
        __builtin_amdgcn_global_load_async_to_lds_b128(
            to_global(&pB[(size_t)(n0 + row) * D + k0 + kc]),
            to_lds(&lB[row * LDS_STRIDE + kc]), 0, 0);
    }
#else
    #pragma unroll
    for (int c = tid; c < 512; c += 256) {
        const int row = c >> 2;
        const int kc  = (c & 3) << 3;
        const uint4 v = *(const uint4*)&fA[(size_t)(m0 + row) * D + k0 + kc];
        *(uint4*)&lA[row * LDS_STRIDE + kc] = v;
    }
    #pragma unroll
    for (int c = tid; c < 512; c += 256) {
        const int row = c >> 2;
        const int kc  = (c & 3) << 3;
        const uint4 v = *(const uint4*)&pB[(size_t)(n0 + row) * D + k0 + kc];
        *(uint4*)&lB[row * LDS_STRIDE + kc] = v;
    }
#endif
}

__device__ __forceinline__ void wait_stage_done() {
#if HAVE_ASYNC_LDS
    __builtin_amdgcn_s_wait_asynccnt(0);
#endif
    __syncthreads();
}

__device__ __forceinline__ void compute_step(
    const unsigned short* __restrict__ lA, const unsigned short* __restrict__ lB,
    v8f acc[2][4], int waveM, int waveN, int lrow, int lhalf)
{
    v16bf aF[2], bF[4];
    #pragma unroll
    for (int mt = 0; mt < 2; ++mt) {
        const int row = waveM * 32 + mt * 16 + lrow;
        const ushort8 lo = *(const ushort8*)&lA[row * LDS_STRIDE + lhalf * 8];
        const ushort8 hi = *(const ushort8*)&lA[row * LDS_STRIDE + 16 + lhalf * 8];
        aF[mt] = cat_bf16(lo, hi);
    }
    #pragma unroll
    for (int nt = 0; nt < 4; ++nt) {
        const int col = waveN * 64 + nt * 16 + lrow;
        const ushort8 lo = *(const ushort8*)&lB[col * LDS_STRIDE + lhalf * 8];
        const ushort8 hi = *(const ushort8*)&lB[col * LDS_STRIDE + 16 + lhalf * 8];
        bF[nt] = cat_bf16(lo, hi);
    }
    #pragma unroll
    for (int mt = 0; mt < 2; ++mt)
        #pragma unroll
        for (int nt = 0; nt < 4; ++nt)
            acc[mt][nt] = __builtin_amdgcn_wmma_f32_16x16x32_bf16(
                false, aF[mt], false, bF[nt], (short)0, acc[mt][nt],
                false, false);
}

__global__ __launch_bounds__(256) void gemm_bf16_kernel(
    const unsigned short* __restrict__ fA, const unsigned short* __restrict__ pB,
    float* __restrict__ sims, int B, int P, int D)
{
    __shared__ unsigned short lA[2][128 * LDS_STRIDE]; // 2 x 10 KB
    __shared__ unsigned short lB[2][128 * LDS_STRIDE]; // 2 x 10 KB

    const int tid   = threadIdx.x;
    const int lane  = tid & 31;
    const int wave  = tid >> 5;
    const int waveM = wave & 3;   // 4 waves along M -> 32 rows each
    const int waveN = wave >> 2;  // 2 waves along N -> 64 cols each
    const int lrow  = lane & 15;
    const int lhalf = lane >> 4;

    const int m0 = blockIdx.y * 128;
    const int n0 = blockIdx.x * 128;

    v8f acc[2][4];
    #pragma unroll
    for (int mt = 0; mt < 2; ++mt)
        #pragma unroll
        for (int nt = 0; nt < 4; ++nt)
            #pragma unroll
            for (int r = 0; r < 8; ++r) acc[mt][nt][r] = 0.f;

    const int nK = D / TILE_K;
    stage_tiles(fA, pB, lA[0], lB[0], m0, n0, 0, D, tid);

    for (int kt = 0; kt < nK; ++kt) {
        wait_stage_done();   // own async loads done + all waves' staging visible
        if (kt + 1 < nK)
            stage_tiles(fA, pB, lA[(kt + 1) & 1], lB[(kt + 1) & 1],
                        m0, n0, (kt + 1) * TILE_K, D, tid);
        compute_step(lA[kt & 1], lB[kt & 1], acc, waveM, waveN, lrow, lhalf);
    }

    // store C tiles (f32 layout: VGPR r -> M = r + 8*lhalf, N = lrow)
    const float invT = 1.0f / 0.07f;
    #pragma unroll
    for (int mt = 0; mt < 2; ++mt)
        #pragma unroll
        for (int nt = 0; nt < 4; ++nt) {
            const int mbase = m0 + waveM * 32 + mt * 16 + lhalf * 8;
            const int n     = n0 + waveN * 64 + nt * 16 + lrow;
            #pragma unroll
            for (int r = 0; r < 8; ++r)
                sims[(size_t)(mbase + r) * P + n] = acc[mt][nt][r] * invT;
        }
}

// ---------------------------------------------------------------------------
// Kernel 2 (fallback, small ws): fused-conversion GEMM, 128x64 block tile.
// ---------------------------------------------------------------------------
__global__ __launch_bounds__(256) void gemm_fused_kernel(
    const float* __restrict__ feat, const float* __restrict__ proxy,
    const float* __restrict__ invn, float* __restrict__ sims,
    int B, int P, int D)
{
    __shared__ unsigned short lA[128 * LDS_STRIDE];
    __shared__ unsigned short lB[64 * LDS_STRIDE];

    const int tid   = threadIdx.x;
    const int lane  = tid & 31;
    const int wave  = tid >> 5;
    const int waveM = wave & 3;
    const int waveN = wave >> 2;
    const int lrow  = lane & 15;
    const int lhalf = lane >> 4;

    const int m0 = blockIdx.y * 128;
    const int n0 = blockIdx.x * 64;

    v8f acc[2][2];
    #pragma unroll
    for (int mt = 0; mt < 2; ++mt)
        #pragma unroll
        for (int nt = 0; nt < 2; ++nt)
            #pragma unroll
            for (int r = 0; r < 8; ++r) acc[mt][nt][r] = 0.f;

    for (int k0 = 0; k0 < D; k0 += TILE_K) {
        for (int c = tid; c < 1024; c += 256) {
            const int row = c >> 3;
            const int kc  = (c & 7) << 2;
            const float4 v = *(const float4*)&feat[(size_t)(m0 + row) * D + k0 + kc];
            const float s = invn[m0 + row];
            unsigned short* dst = &lA[row * LDS_STRIDE + kc];
            dst[0] = f32_to_bf16(v.x * s); dst[1] = f32_to_bf16(v.y * s);
            dst[2] = f32_to_bf16(v.z * s); dst[3] = f32_to_bf16(v.w * s);
        }
        for (int c = tid; c < 512; c += 256) {
            const int row = c >> 3;
            const int kc  = (c & 7) << 2;
            const float4 v = *(const float4*)&proxy[(size_t)(n0 + row) * D + k0 + kc];
            unsigned short* dst = &lB[row * LDS_STRIDE + kc];
            dst[0] = f32_to_bf16(v.x); dst[1] = f32_to_bf16(v.y);
            dst[2] = f32_to_bf16(v.z); dst[3] = f32_to_bf16(v.w);
        }
        __syncthreads();

        v16bf aF[2], bF[2];
        #pragma unroll
        for (int mt = 0; mt < 2; ++mt) {
            const int row = waveM * 32 + mt * 16 + lrow;
            aF[mt] = cat_bf16(*(const ushort8*)&lA[row * LDS_STRIDE + lhalf * 8],
                              *(const ushort8*)&lA[row * LDS_STRIDE + 16 + lhalf * 8]);
        }
        #pragma unroll
        for (int nt = 0; nt < 2; ++nt) {
            const int col = waveN * 32 + nt * 16 + lrow;
            bF[nt] = cat_bf16(*(const ushort8*)&lB[col * LDS_STRIDE + lhalf * 8],
                              *(const ushort8*)&lB[col * LDS_STRIDE + 16 + lhalf * 8]);
        }
        #pragma unroll
        for (int mt = 0; mt < 2; ++mt)
            #pragma unroll
            for (int nt = 0; nt < 2; ++nt)
                acc[mt][nt] = __builtin_amdgcn_wmma_f32_16x16x32_bf16(
                    false, aF[mt], false, bF[nt], (short)0, acc[mt][nt],
                    false, false);
        __syncthreads();
    }

    const float invT = 1.0f / 0.07f;
    #pragma unroll
    for (int mt = 0; mt < 2; ++mt)
        #pragma unroll
        for (int nt = 0; nt < 2; ++nt) {
            const int mbase = m0 + waveM * 32 + mt * 16 + lhalf * 8;
            const int n     = n0 + waveN * 32 + nt * 16 + lrow;
            #pragma unroll
            for (int r = 0; r < 8; ++r)
                sims[(size_t)(mbase + r) * P + n] = acc[mt][nt][r] * invT;
        }
}

// ---------------------------------------------------------------------------
// Kernel 3: per-row loss. One 256-thread block per row. Deterministic.
// ---------------------------------------------------------------------------
__global__ __launch_bounds__(256) void row_loss_kernel(
    const float* __restrict__ sims,
    const int* __restrict__ targets, const int* __restrict__ cams,
    const int* __restrict__ pids, const int* __restrict__ cids,
    float* __restrict__ row_loss, int P)
{
    __shared__ float mneg[8192];
    __shared__ float sVal[256];
    __shared__ int   sIdx[256];
    __shared__ float sRed[256];
    __shared__ int   sCnt[256];
    __shared__ float negVals[NUM_HARDS];

    const int b = blockIdx.x, tid = threadIdx.x;
    const int tgt = targets[b], cam = cams[b];
    const float* row = sims + (size_t)b * P;

    float posSum = 0.f, posMax = NEG_INF; int posCnt = 0;
    for (int j = tid; j < P; j += 256) {
        const float v = row[j];
        const int pid = pids[j], cid = cids[j];
        const bool pos = (pid == tgt) && (cid != cam);
        if (pos) { posSum += v; posCnt++; posMax = fmaxf(posMax, v); }
        mneg[j] = (pid != tgt) ? v : NEG_INF;
    }
    sRed[tid] = posSum; sCnt[tid] = posCnt; sVal[tid] = posMax;
    __syncthreads();
    for (int s = 128; s > 0; s >>= 1) {
        if (tid < s) {
            sRed[tid] += sRed[tid + s];
            sCnt[tid] += sCnt[tid + s];
            sVal[tid]  = fmaxf(sVal[tid], sVal[tid + s]);
        }
        __syncthreads();
    }
    const float totPosSum = sRed[0];
    const int   totPosCnt = sCnt[0];
    const float totPosMax = sVal[0];
    __syncthreads();

    for (int it = 0; it < NUM_HARDS; ++it) {
        float bv = NEG_INF; int bi = 0x7fffffff;
        for (int j = tid; j < P; j += 256) {
            const float v = mneg[j];
            if (v > bv || (v == bv && j < bi)) { bv = v; bi = j; }
        }
        sVal[tid] = bv; sIdx[tid] = bi;
        __syncthreads();
        for (int s = 128; s > 0; s >>= 1) {
            if (tid < s) {
                const float ov = sVal[tid + s]; const int oi = sIdx[tid + s];
                if (ov > sVal[tid] || (ov == sVal[tid] && oi < sIdx[tid])) {
                    sVal[tid] = ov; sIdx[tid] = oi;
                }
            }
            __syncthreads();
        }
        if (tid == 0) {
            negVals[it] = sVal[0];
            if (sIdx[0] < P) mneg[sIdx[0]] = NEG_INF;
        }
        __syncthreads();
    }

    const float M = fmaxf(totPosMax, negVals[0]);
    float es = 0.f;
    for (int j = tid; j < P; j += 256) {
        const int pid = pids[j], cid = cids[j];
        if ((pid == tgt) && (cid != cam)) es += __expf(row[j] - M);
    }
    sRed[tid] = es;
    __syncthreads();
    for (int s = 128; s > 0; s >>= 1) {
        if (tid < s) sRed[tid] += sRed[tid + s];
        __syncthreads();
    }
    if (tid == 0) {
        float tot = sRed[0];
        for (int it = 0; it < NUM_HARDS; ++it) {
            const float v = negVals[it];
            if (v > -1.0e38f) tot += __expf(v - M);
        }
        const float lse = M + __logf(tot);
        row_loss[b] = (totPosCnt > 0) ? (lse - totPosSum / (float)totPosCnt) : 0.f;
    }
}

// ---------------------------------------------------------------------------
// Kernel 4: deterministic final reduction -> out[0] = sum(row_loss)/B
// ---------------------------------------------------------------------------
__global__ __launch_bounds__(256) void final_sum_kernel(
    const float* __restrict__ row_loss, float* __restrict__ out, int B)
{
    __shared__ float s[256];
    const int tid = threadIdx.x;
    float a = 0.f;
    for (int j = tid; j < B; j += 256) a += row_loss[j];
    s[tid] = a; __syncthreads();
    for (int st = 128; st > 0; st >>= 1) {
        if (tid < st) s[tid] += s[tid + st];
        __syncthreads();
    }
    if (tid == 0) out[0] = s[0] / (float)B;
}

// ---------------------------------------------------------------------------
static inline size_t align256(size_t x) { return (x + 255) & ~(size_t)255; }

extern "C" void kernel_launch(void* const* d_in, const int* in_sizes, int n_in,
                              void* d_out, int out_size, void* d_ws, size_t ws_size,
                              hipStream_t stream) {
    const float* feat    = (const float*)d_in[0];
    const float* proxy   = (const float*)d_in[1];
    const int*   targets = (const int*)d_in[2];
    const int*   cams    = (const int*)d_in[3];
    const int*   pids    = (const int*)d_in[4];
    const int*   cids    = (const int*)d_in[5];

    const int B = in_sizes[2];              // targets: [B]
    const int P = in_sizes[4];              // pids:    [P]
    const int D = in_sizes[0] / B;          // features:[B,D]
    float* out = (float*)d_out;

    // workspace layout: invnorm[B] | sims[B*P] | row_loss[B] | fA bf16 | pB bf16
    char* ws = (char*)d_ws;
    size_t off = 0;
    float* invn = (float*)(ws + off);          off += align256((size_t)B * 4);
    float* sims = (float*)(ws + off);          off += align256((size_t)B * P * 4);
    float* row_loss = (float*)(ws + off);      off += align256((size_t)B * 4);
    unsigned short* fA = (unsigned short*)(ws + off); off += align256((size_t)B * D * 2);
    unsigned short* pB = (unsigned short*)(ws + off); off += align256((size_t)P * D * 2);
    const bool bf16_path = (off <= ws_size);

    invnorm_kernel<<<B, 256, 0, stream>>>(feat, invn, D);

    if (bf16_path) {
        convA_kernel<<<B, 256, 0, stream>>>(feat, invn, fA, D);
        const long n4 = (long)P * D / 4;
        int nb = (int)((n4 + 255) / 256); if (nb > 4096) nb = 4096;
        convB_kernel<<<nb, 256, 0, stream>>>(proxy, pB, n4);
        dim3 g(P / 128, B / 128);
        gemm_bf16_kernel<<<g, 256, 0, stream>>>(fA, pB, sims, B, P, D);
    } else {
        dim3 g(P / 64, B / 128);
        gemm_fused_kernel<<<g, 256, 0, stream>>>(feat, proxy, invn, sims, B, P, D);
    }

    row_loss_kernel<<<B, 256, 0, stream>>>(sims, targets, cams, pids, cids,
                                           row_loss, P);

    final_sum_kernel<<<1, 256, 0, stream>>>(row_loss, out, B);
}
